// LSTMModel_45921790329164
// MI455X (gfx1250) — matrix operand order repaired and checked by reading, not compiled, not run on hardware
//
#include <hip/hip_runtime.h>
#include <hip/hip_bf16.h>

typedef __attribute__((ext_vector_type(16))) __bf16 v16bf;
typedef __attribute__((ext_vector_type(8)))  float  v8f;

#define BSZ   8192
#define TLEN  256
#define HDIM  80
#define KP    96     // K padded: 80 h-cols + col80 = s_t, col81 = 1.0, rest 0
#define WAVES 4
#define RPW   16     // batch rows per wave

// LDS carve offsets (bytes)
#define OFF_WHH  0                      // bf16 [320][96]       = 61440 (cols 80/81 = W_ih / bias)
#define OFF_H    61440                  // bf16 [WAVES][16][96] = 12288 (cols 80/81 = s_t / 1.0)
#define OFF_SRC  73728                  // f32  [WAVES][16][8]  = 2048
#define SMEM_BYTES 75776

union Frag { uint4 u[2]; v16bf v; };

#if __has_builtin(__builtin_amdgcn_tanhf)
__device__ __forceinline__ float tanh_hw(float x) { return __builtin_amdgcn_tanhf(x); }
#else
__device__ __forceinline__ float tanh_hw(float x) { return 1.f - 2.f / (1.f + __expf(2.f * x)); }
#endif
// accumulator already holds 0.5*pre_act for i/f/o gates (0.5 folded into weights)
__device__ __forceinline__ float sigm_scaled(float xh) { return __builtin_fmaf(tanh_hw(xh), 0.5f, 0.5f); }

// Load the 12 B fragments (4 gates x 3 K-tiles) for hidden chunk c.
__device__ __forceinline__ void load_chunk(Frag* B, const __bf16* whh, int c, int ln, int half) {
    #pragma unroll
    for (int g = 0; g < 4; ++g) {
        const __bf16* rowp = whh + (g * 80 + c * 16 + ln) * KP + half * 8;
        #pragma unroll
        for (int kt = 0; kt < 3; ++kt) {
            const __bf16* bp = rowp + kt * 32;
            B[g * 3 + kt].u[0] = *(const uint4*)(bp);
            B[g * 3 + kt].u[1] = *(const uint4*)(bp + 16);
        }
    }
}

extern "C" __global__ __launch_bounds__(WAVES * 32)
void lstm_wmma_kernel(const float* __restrict__ src,  const float* __restrict__ W_ih,
                      const float* __restrict__ W_hh, const float* __restrict__ b_ih,
                      const float* __restrict__ b_hh, const float* __restrict__ W_lin,
                      const float* __restrict__ b_lin, float* __restrict__ out)
{
    extern __shared__ char smem[];
    __bf16* whh  = (__bf16*)(smem + OFF_WHH);
    __bf16* hbuf = (__bf16*)(smem + OFF_H);
    float*  srcs = (float*)(smem + OFF_SRC);

    const int tid  = threadIdx.x;
    const int lane = tid & 31;
    const int wave = tid >> 5;
    const int half = lane >> 4;   // 0: lanes 0-15, 1: lanes 16-31
    const int ln   = lane & 15;

    // ---- stage augmented W_hh image: [W_hh | W_ih | b_ih+b_hh | 0...] as bf16.
    //      Rows for i/f/o gates (r<160 or r>=240) are pre-scaled by 0.5 so the
    //      WMMA accumulator directly feeds sigmoid-as-tanh: 0.5 + 0.5*tanh(acc).
    for (int i = tid; i < 320 * KP; i += blockDim.x) {
        int r = i / KP, c = i % KP;
        float v;
        if      (c < HDIM)      v = W_hh[r * HDIM + c];
        else if (c == HDIM)     v = W_ih[r];
        else if (c == HDIM + 1) v = b_ih[r] + b_hh[r];
        else                    v = 0.f;
        if (r < 160 || r >= 240) v *= 0.5f;   // i, f, o gates
        whh[i] = (__bf16)v;
    }
    // h tile: zeros, except augmented col 81 = 1.0 (bias multiplier)
    for (int i = tid; i < WAVES * RPW * KP; i += blockDim.x) {
        int c = i % KP;
        hbuf[i] = (__bf16)((c == HDIM + 1) ? 1.f : 0.f);
    }
    __syncthreads();

    __bf16* hw = hbuf + wave * RPW * KP;   // wave-private augmented h tile [16][96]
    float*  ss = srcs + wave * RPW * 8;    // wave-private src staging [16][8]
    const int rowBase = (blockIdx.x * WAVES + wave) * RPW;

    float creg[5][8];                      // cell state, D-tile layout
    #pragma unroll
    for (int c = 0; c < 5; ++c)
        #pragma unroll
        for (int r = 0; r < 8; ++r) creg[c][r] = 0.f;

    Frag Bbuf[2][12];                      // double-buffered B fragments (time-invariant)
    load_chunk(Bbuf[0], whh, 0, ln, half);

    for (int t = 0; t < TLEN; ++t) {
        if (lane < 16) {
            // refill src staging every 8 steps (coalesced 32B per row)
            if ((t & 7) == 0) {
                const float* sp = src + (size_t)(rowBase + lane) * TLEN + t;
                *(uint4*)(ss + lane * 8)     = *(const uint4*)(sp);
                *(uint4*)(ss + lane * 8 + 4) = *(const uint4*)(sp + 4);
            }
            // place s_t into augmented column 80 of this row
            hw[lane * KP + HDIM] = (__bf16)ss[lane * 8 + (t & 7)];
        }

        // A fragments: augmented h (16 x 96 bf16), ISA 16-bit A layout
        Frag a0, a1, a2;
        {
            const __bf16* hp = hw + ln * KP + half * 8;
            a0.u[0] = *(const uint4*)(hp);            a0.u[1] = *(const uint4*)(hp + 16);
            a1.u[0] = *(const uint4*)(hp + 32);       a1.u[1] = *(const uint4*)(hp + 48);
            a2.u[0] = *(const uint4*)(hp + 64);       a2.u[1] = *(const uint4*)(hp + 80);
        }

        #pragma unroll
        for (int c = 0; c < 5; ++c) {
            Frag* cur = Bbuf[c & 1];
            Frag* nxt = Bbuf[(c & 1) ^ 1];
            // prefetch B for the next chunk (chunk 0 of next step when c==4);
            // in-order same-wave LDS means waits below only cover this prefetch
            load_chunk(nxt, whh, (c == 4) ? 0 : (c + 1), ln, half);

            v8f ai = {}, af = {}, ag = {}, ao = {};
            #pragma unroll
            for (int kt = 0; kt < 3; ++kt) {
                const v16bf av = (kt == 0) ? a0.v : (kt == 1) ? a1.v : a2.v;
                ai = __builtin_amdgcn_wmma_f32_16x16x32_bf16(false, av, false, cur[0 * 3 + kt].v, (short)0, ai, false, false);
                af = __builtin_amdgcn_wmma_f32_16x16x32_bf16(false, av, false, cur[1 * 3 + kt].v, (short)0, af, false, false);
                ag = __builtin_amdgcn_wmma_f32_16x16x32_bf16(false, av, false, cur[2 * 3 + kt].v, (short)0, ag, false, false);
                ao = __builtin_amdgcn_wmma_f32_16x16x32_bf16(false, av, false, cur[3 * 3 + kt].v, (short)0, ao, false, false);
            }
            // acc holds 0.5*pre_act (i/f/o) or pre_act (g); x-proj + bias folded into K
            #pragma unroll
            for (int r = 0; r < 8; ++r) {
                const float gi = sigm_scaled(ai[r]);
                const float gf = sigm_scaled(af[r]);
                const float gg = tanh_hw(ag[r]);
                const float go = sigm_scaled(ao[r]);
                const float cn = gf * creg[c][r] + gi * gg;
                creg[c][r] = cn;
                const float hn = go * tanh_hw(cn);
                hw[(r + half * 8) * KP + c * 16 + ln] = (__bf16)hn;  // cols 82..95 stay 0
            }
        }
    }

    // ---- linear head: out[b] = h_T . W_lin + b_lin ----
    if (lane < 16) {
        float acc = b_lin[0];
        for (int k = 0; k < HDIM; ++k)
            acc += (float)hw[lane * KP + k] * W_lin[k];
        out[rowBase + lane] = acc;
    }
}

extern "C" void kernel_launch(void* const* d_in, const int* in_sizes, int n_in,
                              void* d_out, int out_size, void* d_ws, size_t ws_size,
                              hipStream_t stream) {
    (void)in_sizes; (void)n_in; (void)out_size; (void)d_ws; (void)ws_size;
    const float* src   = (const float*)d_in[0];
    const float* W_ih  = (const float*)d_in[1];
    const float* W_hh  = (const float*)d_in[2];
    const float* b_ih  = (const float*)d_in[3];
    const float* b_hh  = (const float*)d_in[4];
    const float* W_lin = (const float*)d_in[5];
    const float* b_lin = (const float*)d_in[6];
    float* out = (float*)d_out;

    (void)hipFuncSetAttribute((const void*)lstm_wmma_kernel,
                              hipFuncAttributeMaxDynamicSharedMemorySize, SMEM_BYTES);

    dim3 grid(BSZ / (WAVES * RPW));   // 128 blocks
    dim3 block(WAVES * 32);           // 4 waves
    hipLaunchKernelGGL(lstm_wmma_kernel, grid, block, SMEM_BYTES, stream,
                       src, W_ih, W_hh, b_ih, b_hh, W_lin, b_lin, out);
}
